// PairClassifier_2310692405722
// MI455X (gfx1250) — matrix operand order repaired and checked by reading, not compiled, not run on hardware
//
#include <hip/hip_runtime.h>

#define HH 128  // hidden dim (fixed by the model)
#define RT 4    // row tiles (of 16) per wave in the WMMA GEMM

typedef __attribute__((ext_vector_type(16))) _Float16 v16h;
typedef __attribute__((ext_vector_type(8)))  _Float16 v8h;
typedef __attribute__((ext_vector_type(4)))  _Float16 v4h;
typedef __attribute__((ext_vector_type(8)))  float    v8f;

// ---------------------------------------------------------------- utilities

__global__ void fill_f32_kernel(float* p, float v, long long n) {
    long long i = (long long)blockIdx.x * blockDim.x + threadIdx.x;
    if (i < n) p[i] = v;
}

// Bt[n*K + k] = (f16) W[k*Ncol + n]   (transpose + convert, coalesced on Bt)
__global__ void transpose_f16_kernel(const float* __restrict__ W,
                                     _Float16* __restrict__ Bt,
                                     int K, int Ncol) {
    int idx = blockIdx.x * blockDim.x + threadIdx.x;
    if (idx >= K * Ncol) return;
    int n = idx / K;
    int k = idx - n * K;
    Bt[idx] = (_Float16)W[k * Ncol + n];
}

// ---------------------------------------------------------------- graph prep

__global__ void degree_kernel(const int* __restrict__ dst, float* deg, int E) {
    int e = blockIdx.x * blockDim.x + threadIdx.x;
    if (e < E) atomicAdd(&deg[dst[e]], 1.0f);
}

// in-place: deg -> dinv = rsqrt(deg + 1)   (the +1 is the self-loop)
__global__ void dinv_kernel(float* deg, int n) {
    int i = blockIdx.x * blockDim.x + threadIdx.x;
    if (i < n) deg[i] = rsqrtf(deg[i] + 1.0f);
}

// h16[n][j] = relu(x[n] * W_in[j] + b_in[j])   (x is [N,1])
__global__ void input_layer_kernel(const float* __restrict__ x,
                                   const float* __restrict__ Win,
                                   const float* __restrict__ bin,
                                   _Float16* __restrict__ h16, long long total) {
    long long idx = (long long)blockIdx.x * blockDim.x + threadIdx.x;
    if (idx >= total) return;
    int j = (int)(idx & (HH - 1));
    long long node = idx >> 7;
    float v = x[node] * Win[j] + bin[j];
    v = v > 0.0f ? v : 0.0f;
    h16[idx] = (_Float16)v;
}

// -------------------------------------------------------------- WMMA GEMM
// C[M,128] = A[M,KK](f16) @ B[KK,128], B supplied transposed: Bt[128][KK] f16.
// Each wave owns one 16-wide column tile and RT=4 row tiles (64 rows/block),
// sharing one B fragment per k-step across 4 back-to-back WMMAs.
// USE_LDS: Bt staged once per block into LDS via the CDNA5 async global->LDS
// path (GLOBAL_LOAD_ASYNC_TO_LDS_B128 + s_wait_asynccnt), padded stride.
// EMODE 0 (conv):     C32 = v ; Cscaled = v * rowscale[row]^2  (GCN self-loop)
// EMODE 1 (mlp):      C32 = relu(v + bias[col])
template <int KK, bool USE_LDS, int EMODE>
__global__ __launch_bounds__(256)
void wmma_gemm_kernel(const _Float16* __restrict__ A,
                      const _Float16* __restrict__ Bt,
                      const float* __restrict__ bias,
                      const float* __restrict__ rowscale,
                      float* __restrict__ C32,
                      float* __restrict__ Cscaled) {
    extern __shared__ _Float16 ldsB[];            // [HH][KK + 16] when USE_LDS
    constexpr int PADK = KK + 16;                 // 288B col stride: bank-friendly

    if (USE_LDS) {
        // cooperative async stage of Bt (HH*KK halfs) in 16-byte chunks
        constexpr int chunks = (HH * KK) / 8;     // 8 halfs per chunk
        constexpr int kc8 = KK / 8;
        for (int c = threadIdx.x; c < chunks; c += 256) {
            int col = c / kc8;
            int kc  = (c - col * kc8) * 8;
            unsigned lds_addr = (unsigned)(uintptr_t)&ldsB[col * PADK + kc];
            const _Float16* g = Bt + (size_t)col * KK + kc;
            asm volatile("global_load_async_to_lds_b128 %0, %1, off"
                         :: "v"(lds_addr), "v"(g) : "memory");
        }
        asm volatile("s_wait_asynccnt 0x0" ::: "memory");
        __syncthreads();
    }

    const int lane    = threadIdx.x & 31;
    const int wave    = threadIdx.x >> 5;         // col tile 0..7
    const int row0    = blockIdx.x * (16 * RT);   // row tile base (64 rows)
    const int halfSel = (lane < 16) ? 0 : 1;      // wave32 half
    const int mrow    = lane & 15;
    const int ncol    = wave * 16 + mrow;         // C/B column this lane owns

    const _Float16* __restrict__ arowp[RT];
#pragma unroll
    for (int t = 0; t < RT; ++t)
        arowp[t] = A + (size_t)(row0 + t * 16 + mrow) * KK;
    const _Float16* __restrict__ bcolp = Bt + (size_t)ncol * KK;
    const _Float16* __restrict__ blds  = ldsB + (size_t)ncol * PADK;

    v8f acc[RT];
#pragma unroll
    for (int t = 0; t < RT; ++t) acc[t] = (v8f){};

    for (int ks = 0; ks < KK; ks += 32) {
        // B fragment: 16 contiguous K values per lane (lanes 0-15: ks..ks+15,
        // lanes 16-31: ks+16..ks+31), column = lane%16.
        v16h bfrag;
        if (USE_LDS) {
            const _Float16* p = blds + ks + halfSel * 16;   // 16B aligned
            v8h b0 = *(const v8h*)p;
            v8h b1 = *(const v8h*)(p + 8);
#pragma unroll
            for (int i = 0; i < 8; ++i) { bfrag[i] = b0[i]; bfrag[8 + i] = b1[i]; }
        } else {
            bfrag = *(const v16h*)(bcolp + ks + halfSel * 16);
        }

        // A fragments: lanes 0-15 -> K {ks..ks+15},
        //              lanes 16-31 -> K {ks+16..ks+31}
        const int ka = ks + halfSel * 16;
        v16h afrag[RT];
#pragma unroll
        for (int t = 0; t < RT; ++t) {
            v8h alo = *(const v8h*)(arowp[t] + ka);
            v8h ahi = *(const v8h*)(arowp[t] + ka + 8);
#pragma unroll
            for (int i = 0; i < 8; ++i) { afrag[t][i] = alo[i]; afrag[t][8 + i] = ahi[i]; }
        }
#pragma unroll
        for (int t = 0; t < RT; ++t)
            acc[t] = __builtin_amdgcn_wmma_f32_16x16x32_f16(
                false, afrag[t], false, bfrag, (short)0, acc[t], false, false);
    }

    const float bv = (EMODE == 1) ? bias[ncol] : 0.0f;
#pragma unroll
    for (int t = 0; t < RT; ++t) {
        const int rbase = row0 + t * 16 + halfSel * 8;
#pragma unroll
        for (int r = 0; r < 8; ++r) {
            const int row = rbase + r;            // C layout: VGPR r -> M=r / r+8
            const size_t o = (size_t)row * HH + ncol;
            if (EMODE == 1) {
                float v = acc[t][r] + bv;
                C32[o] = v > 0.0f ? v : 0.0f;
            } else {
                const float v = acc[t][r];
                const float s = rowscale[row];
                C32[o] = v;
                Cscaled[o] = v * s * s;
            }
        }
    }
}

// ------------------------------------------------------------- conv pieces

// agg[dst][j..j+3] += xw[src][j..j+3] * dinv[src]*dinv[dst]   (float4 per thread)
__global__ void edge_scatter_kernel(const int* __restrict__ src,
                                    const int* __restrict__ dst,
                                    const float* __restrict__ dinv,
                                    const float* __restrict__ xw,
                                    float* __restrict__ agg, long long total) {
    long long idx = (long long)blockIdx.x * blockDim.x + threadIdx.x;
    if (idx >= total) return;                     // total = E*32
    int q = (int)(idx & 31);
    long long e = idx >> 5;
    int s = src[e], d = dst[e];
    float coef = dinv[s] * dinv[d];
    const float4 v = *(const float4*)(xw + (size_t)s * HH + q * 4);
    float* base = agg + (size_t)d * HH + q * 4;
    atomicAdd(base + 0, v.x * coef);
    atomicAdd(base + 1, v.y * coef);
    atomicAdd(base + 2, v.z * coef);
    atomicAdd(base + 3, v.w * coef);
}

// h16 = relu(agg + b)  (f16 for next GEMM A operand)
__global__ void bias_relu_f16_kernel(const float* __restrict__ agg,
                                     const float* __restrict__ b,
                                     _Float16* __restrict__ h16, long long total) {
    long long idx = (long long)blockIdx.x * blockDim.x + threadIdx.x;
    if (idx >= total) return;
    int j = (int)(idx & (HH - 1));
    float v = agg[idx] + b[j];
    h16[idx] = (_Float16)(v > 0.0f ? v : 0.0f);
}

// --------------------------------------------------------------- pooling

__global__ void pool_kernel(const _Float16* __restrict__ h16,
                            const int* __restrict__ batch,
                            float* __restrict__ pooled,
                            float* __restrict__ counts, long long total) {
    long long idx = (long long)blockIdx.x * blockDim.x + threadIdx.x;
    if (idx >= total) return;                     // total = N*32
    int q = (int)(idx & 31);
    long long node = idx >> 5;
    int g = batch[node];
    const v4h hv = *(const v4h*)(h16 + (size_t)node * HH + q * 4);
    float* base = pooled + (size_t)g * HH + q * 4;
    atomicAdd(base + 0, (float)hv[0]);
    atomicAdd(base + 1, (float)hv[1]);
    atomicAdd(base + 2, (float)hv[2]);
    atomicAdd(base + 3, (float)hv[3]);
    if (q == 0) atomicAdd(&counts[g], 1.0f);
}

__global__ void pool_div_f16_kernel(const float* __restrict__ pooled,
                                    const float* __restrict__ counts,
                                    _Float16* __restrict__ g16, long long total) {
    long long idx = (long long)blockIdx.x * blockDim.x + threadIdx.x;
    if (idx >= total) return;
    int g = (int)(idx >> 7);
    float c = counts[g];
    c = c < 1.0f ? 1.0f : c;
    g16[idx] = (_Float16)(pooled[idx] / c);
}

// ---------------------------------------------------------- pair MLP head

__global__ void build_feats_kernel(const float* __restrict__ za,
                                   const float* __restrict__ zb,
                                   _Float16* __restrict__ feats, int Gn) {
    int idx = blockIdx.x * blockDim.x + threadIdx.x;
    if (idx >= Gn * HH) return;
    int j = idx & (HH - 1);
    int g = idx >> 7;
    float a = za[idx], b = zb[idx];
    _Float16* f = feats + (size_t)g * (4 * HH);
    f[j]          = (_Float16)a;
    f[HH + j]     = (_Float16)b;
    f[2 * HH + j] = (_Float16)fabsf(a - b);
    f[3 * HH + j] = (_Float16)(a * b);
}

__global__ void final_logits_kernel(const float* __restrict__ hid,
                                    const float* __restrict__ Wm2,
                                    const float* __restrict__ bm2,
                                    float* __restrict__ out) {
    __shared__ float red[HH];
    int g = blockIdx.x, j = threadIdx.x;
    red[j] = hid[(size_t)g * HH + j] * Wm2[j];
    __syncthreads();
    for (int s = HH / 2; s > 0; s >>= 1) {
        if (j < s) red[j] += red[j + s];
        __syncthreads();
    }
    if (j == 0) out[g] = red[0] + bm2[0];
}

// ---------------------------------------------------------------- driver

static inline unsigned int nblk(long long n, int t) {
    return (unsigned int)((n + t - 1) / t);
}

extern "C" void kernel_launch(void* const* d_in, const int* in_sizes, int n_in,
                              void* d_out, int out_size, void* d_ws, size_t ws_size,
                              hipStream_t stream) {
    const int N = in_sizes[0];           // nodes per side
    const int E = in_sizes[1] / 2;       // edges per side
    const int G = out_size;              // graphs per side

    const float* xa      = (const float*)d_in[0];
    const int*   ei_a    = (const int*)  d_in[1];
    const int*   batch_a = (const int*)  d_in[2];
    const float* xb      = (const float*)d_in[3];
    const int*   ei_b    = (const int*)  d_in[4];
    const int*   batch_b = (const int*)  d_in[5];
    const float* W_in  = (const float*)d_in[6];
    const float* b_in  = (const float*)d_in[7];
    const float* W_c[3]  = {(const float*)d_in[8], (const float*)d_in[10], (const float*)d_in[12]};
    const float* b_c[3]  = {(const float*)d_in[9], (const float*)d_in[11], (const float*)d_in[13]};
    const float* W_out = (const float*)d_in[14];
    const float* b_out = (const float*)d_in[15];
    const float* W_m1  = (const float*)d_in[16];
    const float* b_m1  = (const float*)d_in[17];
    const float* W_m2  = (const float*)d_in[18];
    const float* b_m2  = (const float*)d_in[19];
    float* logits = (float*)d_out;

    // -------- workspace layout (256B aligned slabs) --------
    char* ws = (char*)d_ws;
    size_t off = 0;
    auto take = [&](size_t bytes) -> char* {
        char* p = ws + off;
        off += (bytes + 255) & ~(size_t)255;
        return p;
    };
    _Float16* h16    = (_Float16*)take((size_t)N * HH * 2);
    float*    xw     = (float*)   take((size_t)N * HH * 4);
    float*    agg    = (float*)   take((size_t)N * HH * 4);
    float*    dinv   = (float*)   take((size_t)N * 4);
    float*    pooled = (float*)   take((size_t)G * HH * 4);
    float*    counts = (float*)   take((size_t)G * 4);
    float*    za     = (float*)   take((size_t)G * HH * 4);
    float*    zb     = (float*)   take((size_t)G * HH * 4);
    _Float16* g16    = (_Float16*)take((size_t)G * HH * 2);
    _Float16* feats  = (_Float16*)take((size_t)G * 4 * HH * 2);
    float*    hid    = (float*)   take((size_t)G * HH * 4);
    _Float16* Wct[3];
    for (int i = 0; i < 3; ++i) Wct[i] = (_Float16*)take((size_t)HH * HH * 2);
    _Float16* Woutt = (_Float16*)take((size_t)HH * HH * 2);
    _Float16* Wm1t  = (_Float16*)take((size_t)4 * HH * HH * 2);
    (void)ws_size;

    const size_t ldsBytes128 = (size_t)HH * (HH + 16) * sizeof(_Float16); // 36864B

    // -------- weight transpose+convert (tiny) --------
    for (int i = 0; i < 3; ++i)
        transpose_f16_kernel<<<nblk(HH * HH, 256), 256, 0, stream>>>(W_c[i], Wct[i], HH, HH);
    transpose_f16_kernel<<<nblk(HH * HH, 256), 256, 0, stream>>>(W_out, Woutt, HH, HH);
    transpose_f16_kernel<<<nblk(4 * HH * HH, 256), 256, 0, stream>>>(W_m1, Wm1t, 4 * HH, HH);

    const long long NH = (long long)N * HH;
    const long long GH = (long long)G * HH;
    const long long EQ = (long long)E * 32;   // quad-threads for scatter
    const long long NQ = (long long)N * 32;   // quad-threads for pool

    // -------- per-side encoder --------
    for (int side = 0; side < 2; ++side) {
        const float* x     = side ? xb : xa;
        const int*   src   = (side ? ei_b : ei_a);
        const int*   dst   = src + E;
        const int*   batch = side ? batch_b : batch_a;
        float*       z     = side ? zb : za;

        // degree -> dinv
        fill_f32_kernel<<<nblk(N, 256), 256, 0, stream>>>(dinv, 0.0f, N);
        degree_kernel<<<nblk(E, 256), 256, 0, stream>>>(dst, dinv, E);
        dinv_kernel<<<nblk(N, 256), 256, 0, stream>>>(dinv, N);

        // input layer
        input_layer_kernel<<<nblk(NH, 256), 256, 0, stream>>>(x, W_in, b_in, h16, NH);

        // 3 GCN convs
        for (int l = 0; l < 3; ++l) {
            // xw = h @ W ; agg = xw * dinv^2 (self-loop, also zero-initializes agg)
            wmma_gemm_kernel<HH, true, 0><<<dim3(N / (16 * RT)), 256, ldsBytes128, stream>>>(
                h16, Wct[l], nullptr, dinv, xw, agg);
            edge_scatter_kernel<<<nblk(EQ, 256), 256, 0, stream>>>(src, dst, dinv, xw, agg, EQ);
            bias_relu_f16_kernel<<<nblk(NH, 256), 256, 0, stream>>>(agg, b_c[l], h16, NH);
        }

        // mean pool
        fill_f32_kernel<<<nblk(GH, 256), 256, 0, stream>>>(pooled, 0.0f, GH);
        fill_f32_kernel<<<nblk(G, 256), 256, 0, stream>>>(counts, 0.0f, G);
        pool_kernel<<<nblk(NQ, 256), 256, 0, stream>>>(h16, batch, pooled, counts, NQ);
        pool_div_f16_kernel<<<nblk(GH, 256), 256, 0, stream>>>(pooled, counts, g16, GH);

        // z = relu(g @ W_out + b_out)
        wmma_gemm_kernel<HH, true, 1><<<dim3(G / (16 * RT)), 256, ldsBytes128, stream>>>(
            g16, Woutt, b_out, nullptr, z, nullptr);
    }

    // -------- pair head --------
    build_feats_kernel<<<nblk((long long)G * HH, 256), 256, 0, stream>>>(za, zb, feats, G);
    // K=512: Bt would need >64KB LDS, keep the (L1-cached) global-load path
    wmma_gemm_kernel<4 * HH, false, 1><<<dim3(G / (16 * RT)), 256, 0, stream>>>(
        feats, Wm1t, b_m1, nullptr, hid, nullptr);
    final_logits_kernel<<<G, HH, 0, stream>>>(hid, W_m2, b_m2, logits);
}
